// VAE_24592982736905
// MI455X (gfx1250) — matrix operand order repaired and checked by reading, not compiled, not run on hardware
//
#include <hip/hip_runtime.h>
#include <hip/hip_bf16.h>

#define FEAT 256
#define HID  128
#define NGRAPH 512

typedef __attribute__((ext_vector_type(16))) __bf16        bf16x16;
typedef __attribute__((ext_vector_type(8)))  float         floatx8;
typedef __attribute__((ext_vector_type(8)))  unsigned int  uintx8;

// round-to-nearest-even f32 -> bf16, two at a time packed into a dword
__device__ __forceinline__ unsigned pack2bf(float lo, float hi) {
  unsigned ul = __float_as_uint(lo); ul += 0x7FFFu + ((ul >> 16) & 1u);
  unsigned uh = __float_as_uint(hi); uh += 0x7FFFu + ((uh >> 16) & 1u);
  return (ul >> 16) | (uh & 0xFFFF0000u);
}

// ---------------------------------------------------------------------------
// x (f32, row-major [N,FEAT]) -> bf16 packed dwords (vectorized b128 in/out)
// ---------------------------------------------------------------------------
__global__ void __launch_bounds__(256)
cvt_f32_bf16(const float* __restrict__ in, unsigned* __restrict__ out, int nVec) {
  int i = blockIdx.x * blockDim.x + threadIdx.x;   // one uint4 (8 floats) each
  if (i >= nVec) return;
  float4 a = ((const float4*)in)[i * 2];
  float4 b = ((const float4*)in)[i * 2 + 1];
  uint4 o;
  o.x = pack2bf(a.x, a.y);
  o.y = pack2bf(a.z, a.w);
  o.z = pack2bf(b.x, b.y);
  o.w = pack2bf(b.z, b.w);
  ((uint4*)out)[i] = o;
}

// ---------------------------------------------------------------------------
// Pre-pack W [FEAT x HID f32] into wave32 WMMA B-fragment order:
// frag = ntile*(FEAT/32)+kblk; per frag 32 lanes x 16 halves (contiguous).
// Lane mapping: l16 = col-in-tile, half selects K+0/K+8; e<8 -> K+e, e>=8 -> K+16+(e-8).
// ---------------------------------------------------------------------------
__global__ void __launch_bounds__(256)
pack_w_frag(const float* __restrict__ W, unsigned short* __restrict__ out) {
  int t = blockIdx.x * blockDim.x + threadIdx.x;
  const int total = (HID / 16) * (FEAT / 32) * 32;      // 2048
  if (t >= total) return;
  int lane  = t & 31;
  int frag  = t >> 5;
  int kblk  = frag % (FEAT / 32);
  int ntile = frag / (FEAT / 32);
  int half = lane >> 4, l16 = lane & 15;
  int bcol  = ntile * 16 + l16;
  int kbase = kblk * 32 + half * 8;
  unsigned short* op = out + (size_t)t * 16;
#pragma unroll
  for (int e = 0; e < 16; ++e) {
    unsigned u = __float_as_uint(W[(size_t)(kbase + e + (e & 8)) * HID + bcol]);
    u += 0x7FFFu + ((u >> 16) & 1u);
    op[e] = (unsigned short)(u >> 16);
  }
}

// ---------------------------------------------------------------------------
// Fused dual GEMM: S1 = X@W1, S2 = X@W2. One wave = one 16x16 tile of both.
// Hot loop: 2 b128 A loads + 4 b128 B loads + 2 v_wmma_f32_16x16x32_bf16.
// ---------------------------------------------------------------------------
__global__ void __launch_bounds__(256)
gcn_gemm2_wmma(const unsigned* __restrict__ Xb,     // bf16 pairs [N, FEAT/2]
               const unsigned* __restrict__ W1f,    // packed fragments
               const unsigned* __restrict__ W2f,
               float* __restrict__ S1, float* __restrict__ S2, int nRows) {
  const int lane = threadIdx.x & 31;
  const int wave = blockIdx.x * (blockDim.x >> 5) + (threadIdx.x >> 5);
  const int mTiles  = (nRows + 15) >> 4;
  const int nTilesN = HID / 16;                       // 8
  const int mtile = wave / nTilesN;
  const int ntile = wave % nTilesN;
  if (mtile >= mTiles) return;

  const int half = lane >> 4;
  const int l16  = lane & 15;
  int arow = mtile * 16 + l16;
  if (arow >= nRows) arow = nRows - 1;
  const int bcol = ntile * 16 + l16;

  const uint4*  xp  = (const uint4*)(Xb + (size_t)arow * (FEAT / 2));
  const uintx8* b1p = (const uintx8*)W1f + (size_t)ntile * (FEAT / 32) * 32 + lane;
  const uintx8* b2p = (const uintx8*)W2f + (size_t)ntile * (FEAT / 32) * 32 + lane;

  floatx8 acc1 = {};
  floatx8 acc2 = {};

#pragma unroll
  for (int kblk = 0; kblk < FEAT / 32; ++kblk) {
    const int k0 = kblk * 32;
    uint4 lo = xp[(k0 + half * 8) >> 3];              // halves k0+half*8 .. +7
    uint4 hi = xp[(k0 + half * 8 + 16) >> 3];         // halves +16 .. +23
    uintx8 au = {lo.x, lo.y, lo.z, lo.w, hi.x, hi.y, hi.z, hi.w};
    bf16x16 a  = __builtin_bit_cast(bf16x16, au);
    bf16x16 b1 = __builtin_bit_cast(bf16x16, b1p[kblk * 32]);
    bf16x16 b2 = __builtin_bit_cast(bf16x16, b2p[kblk * 32]);
    acc1 = __builtin_amdgcn_wmma_f32_16x16x32_bf16(false, a, false, b1,
                                                   (short)0, acc1, false, false);
    acc2 = __builtin_amdgcn_wmma_f32_16x16x32_bf16(false, a, false, b2,
                                                   (short)0, acc2, false, false);
  }

  // epilogue: branch-free for full tiles (always true when N % 16 == 0)
  float* s1p = S1 + (size_t)(mtile * 16 + half * 8) * HID + bcol;
  float* s2p = S2 + (size_t)(mtile * 16 + half * 8) * HID + bcol;
  if (mtile * 16 + 16 <= nRows) {
#pragma unroll
    for (int r = 0; r < 8; ++r) {
      s1p[(size_t)r * HID] = acc1[r];
      s2p[(size_t)r * HID] = acc2[r];
    }
  } else {
#pragma unroll
    for (int r = 0; r < 8; ++r) {
      int row = mtile * 16 + half * 8 + r;
      if (row < nRows) {
        S1[(size_t)row * HID + bcol] = acc1[r];
        S2[(size_t)row * HID + bcol] = acc2[r];
      }
    }
  }
}

// ---------------------------------------------------------------------------
// Edge scatter: agg[dst] += support[src] * w.  32 lanes/edge, float4 per lane.
// support + agg (~75MB) are L2-resident (192MB), atomics stay on-chip.
// ---------------------------------------------------------------------------
__global__ void __launch_bounds__(256)
gcn_scatter(const float* __restrict__ S, const int* __restrict__ srcIdx,
            const int* __restrict__ dstIdx, const float* __restrict__ ew,
            float* __restrict__ agg, int nEdges) {
  int t = blockIdx.x * blockDim.x + threadIdx.x;
  int e = t >> 5;
  if (e >= nEdges) return;
  int c = t & 31;
  int s = srcIdx[e];
  int d = dstIdx[e];
  float wt = ew[e];
  float4 v = ((const float4*)(S + (size_t)s * HID))[c];
  float* ap = agg + (size_t)d * HID + c * 4;
  atomicAdd(ap + 0, v.x * wt);
  atomicAdd(ap + 1, v.y * wt);
  atomicAdd(ap + 2, v.z * wt);
  atomicAdd(ap + 3, v.w * wt);
}

// ---------------------------------------------------------------------------
// bias + ReLU + segment-sum pooling (one 128-thread block per node)
// ---------------------------------------------------------------------------
__global__ void __launch_bounds__(128)
gcn_relu_pool(const float* __restrict__ agg, const float* __restrict__ bias,
              const int* __restrict__ batch, float* __restrict__ pool,
              float* __restrict__ cnt, int nNodes) {
  int node = blockIdx.x;
  if (node >= nNodes) return;
  int f = threadIdx.x;
  float v = agg[(size_t)node * HID + f] + bias[f];
  v = v > 0.0f ? v : 0.0f;
  int g = batch[node];
  atomicAdd(pool + (size_t)g * HID + f, v);
  if (cnt != nullptr && f == 0) atomicAdd(cnt + g, 1.0f);
}

// ---------------------------------------------------------------------------
// finalize: mean pools, deterministic Box-Muller eps, z = eps*exp(lv)+mu
// out = [z | mu | logvar], each G*HID
// ---------------------------------------------------------------------------
__device__ __forceinline__ unsigned pcg_hash(unsigned v) {
  unsigned s = v * 747796405u + 2891336453u;
  unsigned w = ((s >> ((s >> 28u) + 4u)) ^ s) * 277803737u;
  return (w >> 22u) ^ w;
}

__global__ void __launch_bounds__(128)
gcn_finalize(const float* __restrict__ pmu, const float* __restrict__ plv,
             const float* __restrict__ cnt, float* __restrict__ out, int nGraphs) {
  int g = blockIdx.x;
  int f = threadIdx.x;
  int idx = g * HID + f;
  float c = cnt[g];
  c = c > 1.0f ? c : 1.0f;
  float mu = pmu[idx] / c;
  float lv = plv[idx] / c;
  unsigned h1 = pcg_hash(0x9E3779B9u ^ (unsigned)idx);
  unsigned h2 = pcg_hash(h1 + 0x85EBCA6Bu);
  float u1 = ((h1 >> 8) + 1.0f) * (1.0f / 16777217.0f);
  float u2 = (h2 >> 8) * (1.0f / 16777216.0f);
  float eps = sqrtf(-2.0f * __logf(u1)) * __cosf(6.28318530718f * u2);
  out[idx] = eps * __expf(lv) + mu;
  out[(size_t)nGraphs * HID + idx] = mu;
  out[(size_t)2 * nGraphs * HID + idx] = lv;
}

// ---------------------------------------------------------------------------
extern "C" void kernel_launch(void* const* d_in, const int* in_sizes, int n_in,
                              void* d_out, int out_size, void* d_ws, size_t ws_size,
                              hipStream_t stream) {
  const float* x     = (const float*)d_in[0];
  const float* ew    = (const float*)d_in[1];
  const float* W1    = (const float*)d_in[2];
  const float* b1    = (const float*)d_in[3];
  const float* W2    = (const float*)d_in[4];
  const float* b2    = (const float*)d_in[5];
  const int*   eidx  = (const int*)d_in[6];
  const int*   batch = (const int*)d_in[7];

  const int N = in_sizes[0] / FEAT;    // 50000
  const int E = in_sizes[6] / 2;       // 800000
  const int G = NGRAPH;

  // workspace layout (float-sized slots, regions 64-slot aligned):
  // s1 | s2 | agg | pmu | plv | cnt | xbf | w1f | w2f
  float* ws  = (float*)d_ws;
  size_t NH  = (size_t)N * HID;
  size_t GH  = (size_t)G * HID;
  float* s1  = ws;
  float* s2  = s1 + NH;
  float* agg = s2 + NH;
  float* pmu = agg + NH;
  float* plv = pmu + GH;
  float* cnt = plv + GH;
  unsigned* xbf = (unsigned*)(cnt + ((G + 63) & ~63));          // N*FEAT/2 dwords
  unsigned* w1f = xbf + (size_t)N * (FEAT / 2);                 // 16384 dwords
  unsigned* w2f = w1f + (FEAT * HID / 2);

  const int* srcIdx = eidx;
  const int* dstIdx = eidx + E;

  // zero atomic targets: agg + pools + cnt
  size_t zeroBytes = (NH + 2 * GH + (size_t)((G + 63) & ~63)) * sizeof(float);
  hipMemsetAsync((void*)agg, 0, zeroBytes, stream);

  // precision prep: x -> bf16 (once), W -> packed WMMA fragments (tiny)
  int nVec = N * FEAT / 8;
  cvt_f32_bf16<<<(nVec + 255) / 256, 256, 0, stream>>>(x, xbf, nVec);
  pack_w_frag<<<8, 256, 0, stream>>>(W1, (unsigned short*)w1f);
  pack_w_frag<<<8, 256, 0, stream>>>(W2, (unsigned short*)w2f);

  const int mTiles = (N + 15) / 16;
  const int waves  = mTiles * (HID / 16);
  const int gemmBlocks = (waves + 7) / 8;
  const int scatBlocks = (E * 32 + 255) / 256;

  gcn_gemm2_wmma<<<gemmBlocks, 256, 0, stream>>>(xbf, w1f, w2f, s1, s2, N);

  // conv1 -> mu pool (+counts)
  gcn_scatter<<<scatBlocks, 256, 0, stream>>>(s1, srcIdx, dstIdx, ew, agg, E);
  gcn_relu_pool<<<N, 128, 0, stream>>>(agg, b1, batch, pmu, cnt, N);

  // re-zero agg, conv2 -> logvar pool
  hipMemsetAsync((void*)agg, 0, NH * sizeof(float), stream);
  gcn_scatter<<<scatBlocks, 256, 0, stream>>>(s2, srcIdx, dstIdx, ew, agg, E);
  gcn_relu_pool<<<N, 128, 0, stream>>>(agg, b2, batch, plv, nullptr, N);

  gcn_finalize<<<G, 128, 0, stream>>>(pmu, plv, cnt, (float*)d_out, G);
}
